// BoundedMemory_36867999269439
// MI455X (gfx1250) — compile-verified, matrix operand order checked by reading, not run on hardware
//
#include <hip/hip_runtime.h>
#include <hip/hip_bf16.h>
#include <math.h>

typedef __attribute__((ext_vector_type(16))) _Float16 v16h;
typedef __attribute__((ext_vector_type(8)))  float    v8f;
typedef __attribute__((ext_vector_type(4)))  float    v4f;

// ---------------- problem constants ----------------
#define BB        32
#define CHUNK     128
#define DMODEL    1024
#define KD        128
#define NSLOTS    8192
#define TOPK      32

// output layout (float element offsets, reference tuple order)
#define OUT_CONTEXT 0            // 32*1024
#define OUT_SCORES  32768        // 32*16384
#define OUT_EK      557056       // 32*8192*128
#define OUT_EV      34111488
#define OUT_ES      67665920     // 32*8192
#define OUT_EP      67928064     // 32
#define OUT_SK      67928096
#define OUT_SV      101482528
#define OUT_SS      135036960
#define OUT_SP      135299104

// workspace layout (float element offsets)
#define WS_REP   0               // 32*1024
#define WS_Q     32768           // 32*128
#define WS_KNEW  36864
#define WS_VNEW  40960
#define WS_VALS  45056           // epi_vals [32*128] then sem_vals [32*128]

// Pack 16 consecutive f32 into the 16-half A/B operand register.
// K-permutation note: element i of lane L is hardware-logical
// k = (i<8 ? i : i+8) + ((L&16) ? 8 : 0); we load physical column
// kt + ((L&16)?16:0) + i for BOTH A and B, which is a consistent
// permutation of the reduction dimension (dot products are invariant).
__device__ __forceinline__ v16h pack16(v4f f0, v4f f1, v4f f2, v4f f3) {
    v16h a;
    a[0]  = (_Float16)f0.x; a[1]  = (_Float16)f0.y; a[2]  = (_Float16)f0.z; a[3]  = (_Float16)f0.w;
    a[4]  = (_Float16)f1.x; a[5]  = (_Float16)f1.y; a[6]  = (_Float16)f1.z; a[7]  = (_Float16)f1.w;
    a[8]  = (_Float16)f2.x; a[9]  = (_Float16)f2.y; a[10] = (_Float16)f2.z; a[11] = (_Float16)f2.w;
    a[12] = (_Float16)f3.x; a[13] = (_Float16)f3.y; a[14] = (_Float16)f3.z; a[15] = (_Float16)f3.w;
    return a;
}

// ---------------- rep = mean(hidden, axis=1) ----------------
__global__ void mean_hidden(const float* __restrict__ hidden, float* __restrict__ rep) {
    int idx = blockIdx.x * blockDim.x + threadIdx.x;       // 32*1024 threads
    int b = idx >> 10, d = idx & 1023;
    const float* h = hidden + (size_t)b * (CHUNK * DMODEL) + d;
    float s = 0.f;
#pragma unroll 8
    for (int c = 0; c < CHUNK; ++c) s += h[(size_t)c * DMODEL];
    rep[idx] = s * (1.f / CHUNK);
}

// ---------------- generic small GEMM: out[M,N] = (X1 (+X2)) @ W + bias ----------------
// one wave (32 threads) per 16x16 output tile; K multiple of 32
template <bool HASX2>
__global__ void gemm_wmma(const float* __restrict__ X1, const float* __restrict__ X2,
                          const float* __restrict__ W,  const float* __restrict__ bias,
                          float* __restrict__ out, int N, int K) {
    int lane = threadIdx.x;                 // blockDim.x == 32
    int n0 = blockIdx.x * 16;
    int m0 = blockIdx.y * 16;
    int m    = lane & 15;
    int base = (lane & 16) ? 16 : 0;
    int col  = n0 + m;
    const float* xrow  = X1 + (size_t)(m0 + m) * K;
    const float* xrow2 = HASX2 ? (X2 + (size_t)(m0 + m) * K) : nullptr;
    v8f c = {};
    for (int kt = 0; kt < K; kt += 32) {
        const v4f* xp = (const v4f*)(xrow + kt + base);
        v4f f0 = xp[0], f1 = xp[1], f2 = xp[2], f3 = xp[3];
        if (HASX2) {
            const v4f* yp = (const v4f*)(xrow2 + kt + base);
            f0 += yp[0]; f1 += yp[1]; f2 += yp[2]; f3 += yp[3];
        }
        v16h a = pack16(f0, f1, f2, f3);
        v16h bm;
#pragma unroll
        for (int i = 0; i < 16; ++i)
            bm[i] = (_Float16)W[(size_t)(kt + base + i) * N + col];
        c = __builtin_amdgcn_wmma_f32_16x16x32_f16(false, a, false, bm,
                                                   (short)0, c, false, false);
    }
    float bvv = bias[col];
    int mbase = m0 + ((lane & 16) ? 8 : 0);
#pragma unroll
    for (int r = 0; r < 8; ++r)
        out[(size_t)(mbase + r) * N + col] = c[r] + bvv;
}

// ---------------- fused: copy keys bank -> out  AND  scores = keys @ q ----------------
// grid(64, 32) block(256): 8 waves, each handles one 16-slot x 128-dim tile.
// The WMMA A-operand loads ARE the copy-source loads: each keys byte is read
// from HBM exactly once; copy stores are non-temporal.
__global__ void bank_scores_copy(const float* __restrict__ keys, const float* __restrict__ q,
                                 float* __restrict__ out_keys, float* __restrict__ all_scores,
                                 int bank_col_off) {
    int b    = blockIdx.y;
    int wave = threadIdx.x >> 5;
    int lane = threadIdx.x & 31;
    int s0   = blockIdx.x * 128 + wave * 16;
    int m    = lane & 15;
    int base = (lane & 16) ? 16 : 0;

    const float* krow = keys + ((size_t)b * NSLOTS + s0 + m) * KD;
    float*       orow = out_keys + ((size_t)b * NSLOTS + s0 + m) * KD;
    const float* qb   = q + (size_t)b * KD;

    v8f c = {};
#pragma unroll
    for (int kt = 0; kt < KD; kt += 32) {
        const v4f* rp = (const v4f*)(krow + kt + base);
        v4f f0 = rp[0], f1 = rp[1], f2 = rp[2], f3 = rp[3];
        v4f* wp = (v4f*)(orow + kt + base);
        __builtin_nontemporal_store(f0, wp + 0);
        __builtin_nontemporal_store(f1, wp + 1);
        __builtin_nontemporal_store(f2, wp + 2);
        __builtin_nontemporal_store(f3, wp + 3);
        v16h a = pack16(f0, f1, f2, f3);
        const v4f* qp = (const v4f*)(qb + kt + base);
        v16h bq = pack16(qp[0], qp[1], qp[2], qp[3]);
        c = __builtin_amdgcn_wmma_f32_16x16x32_f16(false, a, false, bq,
                                                   (short)0, c, false, false);
    }
    // C[mrow, n] = score(slot s0+mrow) replicated across n; lanes with n==0 write out
    if ((lane & 15) == 0) {
        int mbase = (lane & 16) ? 8 : 0;
        float* so = all_scores + (size_t)b * (2 * NSLOTS) + bank_col_off + s0 + mbase;
#pragma unroll
        for (int r = 0; r < 8; ++r) so[r] = c[r];
    }
}

// ---------------- wide streaming copy (non-temporal both sides) ----------------
__global__ void copy_f4(const v4f* __restrict__ src, v4f* __restrict__ dst, size_t n4) {
    size_t i = (size_t)blockIdx.x * blockDim.x + threadIdx.x;
    size_t stride = (size_t)gridDim.x * blockDim.x;
    for (; i < n4; i += stride) {
        v4f v = __builtin_nontemporal_load(src + i);
        __builtin_nontemporal_store(v, dst + i);
    }
}

// ---------------- top-k + softmax + weighted gather, per (bank, batch) ----------------
__global__ void topk_softmax_gather(const float* __restrict__ all_scores,
                                    const float* __restrict__ epi_values,
                                    const float* __restrict__ sem_values,
                                    float* __restrict__ vals_ws) {
    __shared__ float sc[NSLOTS];
    __shared__ float rv[256];
    __shared__ int   ri[256];
    __shared__ float topv[TOPK];
    __shared__ int   topi[TOPK];
    __shared__ float w[TOPK];
    int bank = blockIdx.x, b = blockIdx.y, tid = threadIdx.x;

    const float* src = all_scores + (size_t)b * (2 * NSLOTS) + (size_t)bank * NSLOTS;
    for (int j = tid; j < NSLOTS; j += 256) sc[j] = src[j];
    __syncthreads();

    for (int it = 0; it < TOPK; ++it) {
        float v = -INFINITY; int idx = 0x7fffffff;
        for (int j = tid; j < NSLOTS; j += 256) {
            float x = sc[j];
            if (x > v) { v = x; idx = j; }          // strict > : lowest index wins in-thread
        }
        rv[tid] = v; ri[tid] = idx;
        __syncthreads();
        for (int off = 128; off > 0; off >>= 1) {
            if (tid < off) {
                float v2 = rv[tid + off]; int i2 = ri[tid + off];
                if (v2 > rv[tid] || (v2 == rv[tid] && i2 < ri[tid])) { rv[tid] = v2; ri[tid] = i2; }
            }
            __syncthreads();
        }
        if (tid == 0) { topv[it] = rv[0]; topi[it] = ri[0]; sc[ri[0]] = -INFINITY; }
        __syncthreads();
    }

    if (tid == 0) {
        float mx = topv[0];                          // sorted descending
        float s = 0.f;
        for (int i = 0; i < TOPK; ++i) { float e = __expf(topv[i] - mx); w[i] = e; s += e; }
        float inv = 1.f / s;
        for (int i = 0; i < TOPK; ++i) w[i] *= inv;
    }
    __syncthreads();

    const float* V = bank ? sem_values : epi_values;
    if (tid < KD) {
        float acc = 0.f;
#pragma unroll 8
        for (int i = 0; i < TOPK; ++i)
            acc += w[i] * V[((size_t)b * NSLOTS + topi[i]) * KD + tid];
        vals_ws[(size_t)bank * (BB * KD) + (size_t)b * KD + tid] = acc;
    }
}

// ---------------- masked circular-buffer slot update + ptr ----------------
__global__ void write_fix(const float* __restrict__ k_new, const float* __restrict__ v_new,
                          const float* __restrict__ s_new, const int* __restrict__ ptr,
                          float* __restrict__ outK, float* __restrict__ outV,
                          float* __restrict__ outS, float* __restrict__ outP) {
    int b = blockIdx.x, t = threadIdx.x;             // 128 threads
    float sv = s_new[b];
    bool mask = sv > 0.f;
    unsigned slot = ((unsigned)ptr[b]) % (unsigned)NSLOTS;
    if (mask) {
        outK[((size_t)b * NSLOTS + slot) * KD + t] = k_new[(size_t)b * KD + t];
        outV[((size_t)b * NSLOTS + slot) * KD + t] = v_new[(size_t)b * KD + t];
        if (t == 0) outS[(size_t)b * NSLOTS + slot] = sv;
    }
    if (t == 0) outP[b] = (float)(ptr[b] + (mask ? 1 : 0));
}

extern "C" void kernel_launch(void* const* d_in, const int* in_sizes, int n_in,
                              void* d_out, int out_size, void* d_ws, size_t ws_size,
                              hipStream_t stream) {
    const float* hidden     = (const float*)d_in[0];
    const float* query      = (const float*)d_in[1];
    const float* ws_epi     = (const float*)d_in[2];
    const float* ws_sem     = (const float*)d_in[3];
    const float* epi_keys   = (const float*)d_in[4];
    const float* epi_vals_b = (const float*)d_in[5];
    const float* epi_scores = (const float*)d_in[6];
    const float* sem_keys   = (const float*)d_in[7];
    const float* sem_vals_b = (const float*)d_in[8];
    const float* sem_scores = (const float*)d_in[9];
    const int*   epi_ptr    = (const int*)d_in[10];
    const int*   sem_ptr    = (const int*)d_in[11];
    const float* Wk         = (const float*)d_in[12];
    const float* bk         = (const float*)d_in[13];
    const float* Wv         = (const float*)d_in[14];
    const float* bv         = (const float*)d_in[15];
    const float* Wr         = (const float*)d_in[16];
    const float* br         = (const float*)d_in[17];

    float* out = (float*)d_out;
    float* ws  = (float*)d_ws;

    float* rep   = ws + WS_REP;
    float* qbuf  = ws + WS_Q;
    float* knew  = ws + WS_KNEW;
    float* vnew  = ws + WS_VNEW;
    float* valsb = ws + WS_VALS;     // [2][32][128]

    // 1) rep = mean(hidden, axis=1)
    mean_hidden<<<dim3((BB * DMODEL) / 256), dim3(256), 0, stream>>>(hidden, rep);

    // 2) projections: q, k_new, v_new  (M=32, N=128, K=1024)
    gemm_wmma<false><<<dim3(KD / 16, BB / 16), dim3(32), 0, stream>>>(query, nullptr, Wk, bk, qbuf, KD, DMODEL);
    gemm_wmma<false><<<dim3(KD / 16, BB / 16), dim3(32), 0, stream>>>(rep,   nullptr, Wk, bk, knew, KD, DMODEL);
    gemm_wmma<false><<<dim3(KD / 16, BB / 16), dim3(32), 0, stream>>>(rep,   nullptr, Wv, bv, vnew, KD, DMODEL);

    // 3) fused keys copy + scores (both banks)
    bank_scores_copy<<<dim3(NSLOTS / 128, BB), dim3(256), 0, stream>>>(
        epi_keys, qbuf, out + OUT_EK, out + OUT_SCORES, 0);
    bank_scores_copy<<<dim3(NSLOTS / 128, BB), dim3(256), 0, stream>>>(
        sem_keys, qbuf, out + OUT_SK, out + OUT_SCORES, NSLOTS);

    // 4) values / scores bank copies
    copy_f4<<<dim3(4096), dim3(256), 0, stream>>>((const v4f*)epi_vals_b,
        (v4f*)(out + OUT_EV), (size_t)BB * NSLOTS * KD / 4);
    copy_f4<<<dim3(4096), dim3(256), 0, stream>>>((const v4f*)sem_vals_b,
        (v4f*)(out + OUT_SV), (size_t)BB * NSLOTS * KD / 4);
    copy_f4<<<dim3(64), dim3(256), 0, stream>>>((const v4f*)epi_scores,
        (v4f*)(out + OUT_ES), (size_t)BB * NSLOTS / 4);
    copy_f4<<<dim3(64), dim3(256), 0, stream>>>((const v4f*)sem_scores,
        (v4f*)(out + OUT_SS), (size_t)BB * NSLOTS / 4);

    // 5) top-k + softmax + gather (per bank, per batch)
    topk_softmax_gather<<<dim3(2, BB), dim3(256), 0, stream>>>(
        out + OUT_SCORES, epi_vals_b, sem_vals_b, valsb);

    // 6) context = (epi_vals + sem_vals) @ Wr + br   (M=32, N=1024, K=128)
    gemm_wmma<true><<<dim3(DMODEL / 16, BB / 16), dim3(32), 0, stream>>>(
        valsb, valsb + BB * KD, Wr, br, out + OUT_CONTEXT, DMODEL, KD);

    // 7) masked slot writes + pointers
    write_fix<<<dim3(BB), dim3(KD), 0, stream>>>(knew, vnew, ws_epi, epi_ptr,
        out + OUT_EK, out + OUT_EV, out + OUT_ES, out + OUT_EP);
    write_fix<<<dim3(BB), dim3(KD), 0, stream>>>(knew, vnew, ws_sem, sem_ptr,
        out + OUT_SK, out + OUT_SV, out + OUT_SS, out + OUT_SP);
}
